// Attention_66546223284383
// MI455X (gfx1250) — matrix operand order repaired and checked by reading, not compiled, not run on hardware
//
#include <hip/hip_runtime.h>
#include <hip/hip_bf16.h>
#include <stdint.h>

// ---------------- problem constants ----------------
#define BN    16      // batch
#define NN    1044    // att tokens
#define NP    1056    // padded tokens (66*16)
#define CC    512     // input dim
#define HQ    1024    // qkv rows
#define NH    8       // heads
#define KD    32      // key dim
#define DV    64      // value dim per head
#define DH    512     // NH*DV
#define IMG   1024    // 32x32
#define SSZ   32

typedef __attribute__((ext_vector_type(16))) _Float16 v16h;
typedef __attribute__((ext_vector_type(8)))  float    v8f;

union F16Frag { v16h v; uint32_t u[8]; };

// A-fragment: one row per lane (m = lane%16). k order used here (contiguous, not the ISA table order):
// VGPR i(0..7): K = 16*half + 2i,+1 (matches B-fragment order)
__device__ __forceinline__ v16h load_a_frag(const _Float16* row, int half) {
  F16Frag f;
#pragma unroll
  for (int i = 0; i < 8; ++i) {
    f.u[i]     = *reinterpret_cast<const uint32_t*>(row + 16 * half + 2 * i);
  }
  return f.v;
}

// B-fragment: one column per lane (n = lane%16); rowT points at B^T row n
// (K-contiguous). VGPR j: K = 16*half + 2j,+1 -> 16 consecutive halves.
__device__ __forceinline__ v16h load_b_frag(const _Float16* rowT, int half) {
  F16Frag f;
#pragma unroll
  for (int j = 0; j < 8; ++j)
    f.u[j] = *reinterpret_cast<const uint32_t*>(rowT + 16 * half + 2 * j);
  return f.v;
}

__device__ __forceinline__ v8f wmma16(v16h a, v16h b, v8f c) {
  return __builtin_amdgcn_wmma_f32_16x16x32_f16(false, a, false, b, (short)0, c,
                                                false, false);
}

// ---------------- prep: x -> f16 (padded rows zero) ----------------
__global__ void k_prep_x(const float* __restrict__ x, _Float16* __restrict__ x16) {
  size_t idx = (size_t)blockIdx.x * blockDim.x + threadIdx.x;
  const size_t total = (size_t)BN * NP * CC;
  if (idx >= total) return;
  int c = (int)(idx % CC);
  size_t t = idx / CC;
  int n = (int)(t % NP);
  int b = (int)(t / NP);
  float v = (n < NN) ? x[((size_t)b * NN + n) * CC + c] : 0.0f;
  x16[idx] = (_Float16)v;
}

// ---------------- prep: weights -> f16, BN folding, o16 tail zero ----------
__global__ void k_prep_w(const float* __restrict__ qkv_w,
                         const float* __restrict__ proj_w,
                         const float* __restrict__ out_w,
                         const float* __restrict__ g, const float* __restrict__ be,
                         const float* __restrict__ mu, const float* __restrict__ var,
                         _Float16* __restrict__ wq16, _Float16* __restrict__ wp16,
                         _Float16* __restrict__ wo16,
                         float* __restrict__ bnsc, float* __restrict__ bnsh,
                         _Float16* __restrict__ o16_tail) {
  size_t idx = (size_t)blockIdx.x * blockDim.x + threadIdx.x;
  if (idx < (size_t)HQ * CC)                        // qkv_w 1024x512
    wq16[idx] = (_Float16)qkv_w[idx];
  if (idx < (size_t)IMG * NP) {                     // proj_w 1024x1056 (pad 0)
    int m = (int)(idx / NP), col = (int)(idx % NP);
    wp16[idx] = (_Float16)((col < NN) ? proj_w[(size_t)m * NN + col] : 0.0f);
  }
  if (idx < (size_t)CC * DH)                        // out_w 512x512
    wo16[idx] = (_Float16)out_w[idx];
  if (idx < HQ) {
    float sc = g[idx] * rsqrtf(var[idx] + 1e-5f);
    bnsc[idx] = sc;
    bnsh[idx] = be[idx] - mu[idx] * sc;
  }
  if (idx < 64) o16_tail[idx] = (_Float16)0.0f;     // over-read pad for proj GEMM
}

// ---------------- GEMM1: qkv = x @ qkv_w^T, fused BN + scatter -------------
__global__ __launch_bounds__(256) void k_qkv(
    const _Float16* __restrict__ x16, const _Float16* __restrict__ wq16,
    const float* __restrict__ bnsc, const float* __restrict__ bnsh,
    _Float16* __restrict__ q16, _Float16* __restrict__ k16,
    _Float16* __restrict__ vt16, float* __restrict__ vf32) {
  const int lane = threadIdx.x & 31, wave = threadIdx.x >> 5;
  const int half = lane >> 4, ln = lane & 15;
  const int b = blockIdx.x;
  const int m0 = blockIdx.y * 16;            // token tile (66)
  const int h0 = blockIdx.z * 128 + wave * 16;  // qkv column tile
  const _Float16* arow = x16 + ((size_t)b * NP + m0 + ln) * CC;
  const _Float16* brow = wq16 + (size_t)(h0 + ln) * CC;
  v8f acc = {};
#pragma unroll 4
  for (int kk = 0; kk < CC; kk += 32)
    acc = wmma16(load_a_frag(arow + kk, half), load_b_frag(brow + kk, half), acc);

  const int col = h0 + ln;
  const float sc = bnsc[col], sh = bnsh[col];
  const int head = col >> 7, wo = col & 127;
  const size_t bh = (size_t)(b * NH + head);
#pragma unroll
  for (int r = 0; r < 8; ++r) {
    const int n = m0 + half * 8 + r;
    const float v = acc[r] * sc + sh;
    if (wo < KD) {
      q16[(bh * NP + n) * KD + wo] = (_Float16)(v * 0.17677669529663687f);
    } else if (wo < 2 * KD) {
      k16[(bh * NP + n) * KD + (wo - KD)] = (_Float16)v;
    } else {
      const int dd = wo - 2 * KD;
      vt16[(bh * 64 + dd) * NP + n] = (_Float16)v;   // V^T for PV B-frags
      vf32[(bh * NP + n) * 64 + dd] = v;             // fp32 V for conv branch
    }
  }
}

// ---------------- flash attention: per-wave 16-row tile --------------------
#define ATT_WAVES 2
__global__ __launch_bounds__(ATT_WAVES * 32) void k_attn(
    const _Float16* __restrict__ q16, const _Float16* __restrict__ k16,
    const _Float16* __restrict__ vt16, _Float16* __restrict__ o16) {
  __shared__ __align__(16) _Float16 plds[ATT_WAVES][16][32];
  const int lane = threadIdx.x & 31, wave = threadIdx.x >> 5;
  const int half = lane >> 4, ln = lane & 15;
  const int bh = blockIdx.x;                              // b*8 + h
  const int m0 = (blockIdx.y * ATT_WAVES + wave) * 16;    // query tile
  const _Float16* qrow = q16 + ((size_t)bh * NP + m0 + ln) * KD;
  const v16h aq = load_a_frag(qrow, half);                // K=32 in one frag
  v8f o_acc[4] = {};
  float rmax[8], rsum[8];
#pragma unroll
  for (int r = 0; r < 8; ++r) { rmax[r] = -1e30f; rsum[r] = 0.0f; }
  _Float16* pl = &plds[wave][0][0];
  const v8f vzero = {};

  for (int kt = 0; kt < NP; kt += 32) {
    const _Float16* krow0 = k16 + ((size_t)bh * NP + kt + ln) * KD;
    const v16h bk0 = load_b_frag(krow0, half);
    const v16h bk1 = load_b_frag(krow0 + 16 * KD, half);
    v8f s0 = wmma16(aq, bk0, vzero);
    v8f s1 = wmma16(aq, bk1, vzero);
    const bool ok0 = (kt + ln) < NN;
    const bool ok1 = (kt + 16 + ln) < NN;
    float cf[8];
#pragma unroll
    for (int r = 0; r < 8; ++r) {
      float a0 = ok0 ? s0[r] : -1e30f;
      float a1 = ok1 ? s1[r] : -1e30f;
      float tm = fmaxf(a0, a1);
      tm = fmaxf(tm, __shfl_xor(tm, 1));
      tm = fmaxf(tm, __shfl_xor(tm, 2));
      tm = fmaxf(tm, __shfl_xor(tm, 4));
      tm = fmaxf(tm, __shfl_xor(tm, 8));          // 16-lane halves = C rows 0-7/8-15
      const float nm = fmaxf(rmax[r], tm);
      const float c = __expf(rmax[r] - nm);
      const float p0 = __expf(a0 - nm);
      const float p1 = __expf(a1 - nm);
      float ps = p0 + p1;
      ps += __shfl_xor(ps, 1);
      ps += __shfl_xor(ps, 2);
      ps += __shfl_xor(ps, 4);
      ps += __shfl_xor(ps, 8);
      rsum[r] = rsum[r] * c + ps;
      rmax[r] = nm;
      cf[r] = c;
      pl[(half * 8 + r) * 32 + ln]      = (_Float16)p0;   // C-layout -> LDS
      pl[(half * 8 + r) * 32 + 16 + ln] = (_Float16)p1;
    }
    asm volatile("s_wait_dscnt 0" ::: "memory");          // stores visible
    const v16h ap = load_a_frag(pl + ln * 32, half);      // reload as A-frag
#pragma unroll
    for (int j = 0; j < 4; ++j) {
      v8f t = o_acc[j];
#pragma unroll
      for (int r = 0; r < 8; ++r) t[r] *= cf[r];
      const _Float16* vrow = vt16 + ((size_t)bh * 64 + j * 16 + ln) * NP + kt;
      o_acc[j] = wmma16(ap, load_b_frag(vrow, half), t);
    }
    asm volatile("s_wait_dscnt 0" ::: "memory");          // loads done before reuse
  }

  const int b = bh >> 3, h = bh & 7;
#pragma unroll
  for (int r = 0; r < 8; ++r) {
    const int n = m0 + half * 8 + r;
    if (n >= NN) continue;
    const float inv = 1.0f / rsum[r];
#pragma unroll
    for (int j = 0; j < 4; ++j) {
      const int dd = j * 16 + ln;
      // torch raw flatten: O stored as (B, N, H*d) so proj A-rows are contiguous
      o16[((size_t)b * NN + n) * DH + h * DV + dd] = (_Float16)(o_acc[j][r] * inv);
    }
  }
}

// ---------------- depthwise 3x3 conv branch (hardswish input) --------------
__global__ void k_conv(const float* __restrict__ vf32,
                       const float* __restrict__ conv_w,
                       const float* __restrict__ conv_b,
                       float* __restrict__ vc) {
  const size_t idx = (size_t)blockIdx.x * blockDim.x + threadIdx.x;
  if (idx >= (size_t)BN * DH * IMG) return;
  const int p = (int)(idx & 1023);
  const int ch = (int)((idx >> 10) & 511);
  const int b = (int)(idx >> 19);
  const int y = p >> 5, xx = p & 31;
  float acc = conv_b[ch];
#pragma unroll
  for (int ky = 0; ky < 3; ++ky) {
    const int yy = y + ky - 1;
    if (yy < 0 || yy >= SSZ) continue;
#pragma unroll
    for (int kx = 0; kx < 3; ++kx) {
      const int xc = xx + kx - 1;
      if (xc < 0 || xc >= SSZ) continue;
      // raw reshape (H,IMG,d) -> (DH,32,32)
      const int f = (ch << 10) + (yy << 5) + xc;
      const int h = f >> 16, n = (f >> 6) & 1023, dd = f & 63;
      const float v = vf32[(((size_t)(b * NH + h)) * NP + n) * 64 + dd];
      const float hs = v * fminf(fmaxf(v + 3.0f, 0.0f), 6.0f) * (1.0f / 6.0f);
      acc += hs * conv_w[ch * 9 + ky * 3 + kx];
    }
  }
  vc[idx] = acc;
}

// ---------------- GEMM3: xo = o_r @ proj_w^T + proj_b + vc -----------------
__global__ __launch_bounds__(256) void k_proj(
    const _Float16* __restrict__ o16, const _Float16* __restrict__ wp16,
    const float* __restrict__ proj_b, const float* __restrict__ vc,
    _Float16* __restrict__ xo16) {
  const int lane = threadIdx.x & 31, wave = threadIdx.x >> 5;
  const int half = lane >> 4, ln = lane & 15;
  const int b = blockIdx.x;
  const int dd0 = blockIdx.y * 16;               // 512/16 = 32 tiles
  const int m0 = blockIdx.z * 128 + wave * 16;   // 1024 cols
  // o_r row dd = o16 flat + b*N*DH + dd*N (stride N=1044; pad reads x 0-weights)
  const _Float16* arow = o16 + (size_t)b * NN * DH + (size_t)(dd0 + ln) * NN;
  const _Float16* brow = wp16 + (size_t)(m0 + ln) * NP;
  v8f acc = {};
  for (int kk = 0; kk < NP; kk += 32)
    acc = wmma16(load_a_frag(arow + kk, half), load_b_frag(brow + kk, half), acc);
  const int m = m0 + ln;
  const float pb = proj_b[m];
#pragma unroll
  for (int r = 0; r < 8; ++r) {
    const int dd = dd0 + half * 8 + r;
    const float v = acc[r] + pb + vc[((size_t)b * DH + dd) * IMG + m];
    xo16[((size_t)b * IMG + m) * DH + dd] = (_Float16)v;  // store transposed
  }
}

// ---------------- GEMM4: out = xo^T @ out_w^T + out_b ----------------------
__global__ __launch_bounds__(256) void k_out(
    const _Float16* __restrict__ xo16, const _Float16* __restrict__ wo16,
    const float* __restrict__ out_b, float* __restrict__ out) {
  const int lane = threadIdx.x & 31, wave = threadIdx.x >> 5;
  const int half = lane >> 4, ln = lane & 15;
  const int b = blockIdx.x;
  const int m0 = blockIdx.y * 16;                // 1024/16 = 64 tiles
  const int c0 = blockIdx.z * 128 + wave * 16;   // 512 cols
  const _Float16* arow = xo16 + ((size_t)b * IMG + m0 + ln) * DH;
  const _Float16* brow = wo16 + (size_t)(c0 + ln) * DH;
  v8f acc = {};
#pragma unroll 4
  for (int kk = 0; kk < DH; kk += 32)
    acc = wmma16(load_a_frag(arow + kk, half), load_b_frag(brow + kk, half), acc);
  const int cc = c0 + ln;
  const float ob = out_b[cc];
#pragma unroll
  for (int r = 0; r < 8; ++r) {
    const int m = m0 + half * 8 + r;
    out[((size_t)b * IMG + m) * CC + cc] = acc[r] + ob;
  }
}

// ---------------- host launch ----------------------------------------------
extern "C" void kernel_launch(void* const* d_in, const int* in_sizes, int n_in,
                              void* d_out, int out_size, void* d_ws, size_t ws_size,
                              hipStream_t stream) {
  const float* x      = (const float*)d_in[0];
  const float* qkv_w  = (const float*)d_in[1];
  const float* bn_g   = (const float*)d_in[2];
  const float* bn_b   = (const float*)d_in[3];
  const float* bn_m   = (const float*)d_in[4];
  const float* bn_v   = (const float*)d_in[5];
  const float* conv_w = (const float*)d_in[6];
  const float* conv_b = (const float*)d_in[7];
  const float* proj_w = (const float*)d_in[8];
  const float* proj_b = (const float*)d_in[9];
  const float* out_w  = (const float*)d_in[10];
  const float* out_b  = (const float*)d_in[11];
  float* out = (float*)d_out;

  auto al = [](size_t v) { return (v + 255) & ~(size_t)255; };
  char* p = (char*)d_ws;
  _Float16* x16  = (_Float16*)p; p += al((size_t)BN * NP * CC * 2);
  _Float16* wq16 = (_Float16*)p; p += al((size_t)HQ * CC * 2);
  _Float16* wp16 = (_Float16*)p; p += al((size_t)IMG * NP * 2);
  _Float16* wo16 = (_Float16*)p; p += al((size_t)CC * DH * 2);
  float*    bnsc = (float*)p;    p += al((size_t)HQ * 4);
  float*    bnsh = (float*)p;    p += al((size_t)HQ * 4);
  _Float16* q16  = (_Float16*)p; p += al((size_t)BN * NH * NP * KD * 2);
  _Float16* k16  = (_Float16*)p; p += al((size_t)BN * NH * NP * KD * 2);
  _Float16* vt16 = (_Float16*)p; p += al((size_t)BN * NH * 64 * NP * 2);
  float*    vf32 = (float*)p;    p += al((size_t)BN * NH * NP * 64 * 4);
  _Float16* o16  = (_Float16*)p; p += al(((size_t)BN * NN * DH + 64) * 2);
  _Float16* xo16 = (_Float16*)p; p += al((size_t)BN * IMG * DH * 2);
  float*    vc   = (float*)p;    p += al((size_t)BN * DH * IMG * 4);
  (void)ws_size; (void)n_in; (void)in_sizes; (void)out_size;

  const size_t xtot = (size_t)BN * NP * CC;
  k_prep_x<<<(unsigned)((xtot + 255) / 256), 256, 0, stream>>>(x, x16);

  const size_t wtot = (size_t)IMG * NP;  // largest prep region
  k_prep_w<<<(unsigned)((wtot + 255) / 256), 256, 0, stream>>>(
      qkv_w, proj_w, out_w, bn_g, bn_b, bn_m, bn_v, wq16, wp16, wo16, bnsc, bnsh,
      o16 + (size_t)BN * NN * DH);

  k_qkv<<<dim3(BN, NP / 16, HQ / 128), 256, 0, stream>>>(
      x16, wq16, bnsc, bnsh, q16, k16, vt16, vf32);

  k_attn<<<dim3(BN * NH, (NP / 16) / ATT_WAVES), ATT_WAVES * 32, 0, stream>>>(
      q16, k16, vt16, o16);

  const size_t ctot = (size_t)BN * DH * IMG;
  k_conv<<<(unsigned)((ctot + 255) / 256), 256, 0, stream>>>(vf32, conv_w, conv_b, vc);

  k_proj<<<dim3(BN, DH / 16, IMG / 128), 256, 0, stream>>>(o16, wp16, proj_b, vc, xo16);

  k_out<<<dim3(BN, IMG / 16, CC / 128), 256, 0, stream>>>(xo16, wo16, out_b, out);
}
